// MAFN_59090160058897
// MI455X (gfx1250) — matrix operand order repaired and checked by reading, not compiled, not run on hardware
//
#include <hip/hip_runtime.h>

// ---------------------------------------------------------------------------
// MAFN fused pipeline for MI455X (gfx1250, wave32, WMMA 16x16x32 f16).
// Inputs (setup_inputs order):
//  0 x (B,L,C) f32        1 g_reduce_w (R,C)   2 l_reduce_w (R,C)
//  3 conv3_w (64,64,3,3)  4 conv3_b (64)       5 conv5_w (64,64,5,5)
//  6 conv5_b (64)         7 channel_w (C,R)    8 channel_b (C)
//  9 spatial_w (1,3R)    10 spatial_b (1)     11 star_scale  12 star_bias
// ---------------------------------------------------------------------------

#define B_N 8192
#define L_N 49
#define C_N 512
#define R_N 128

typedef _Float16 half4  __attribute__((ext_vector_type(4)));
typedef _Float16 half8  __attribute__((ext_vector_type(8)));
typedef _Float16 half16 __attribute__((ext_vector_type(16)));
typedef float    floatv4 __attribute__((ext_vector_type(4)));
typedef float    float8  __attribute__((ext_vector_type(8)));

static __device__ __forceinline__ half16 cat8(half8 a, half8 b) {
  return __builtin_shufflevector(a, b, 0, 1, 2, 3, 4, 5, 6, 7,
                                 8, 9, 10, 11, 12, 13, 14, 15);
}

static __device__ __forceinline__ float8 wmma_f16(half16 a, half16 b, float8 c) {
  // D = A(16x32 f16) * B(32x16 f16) + C(16x16 f32)
  return __builtin_amdgcn_wmma_f32_16x16x32_f16(
      /*neg_a=*/false, a, /*neg_b=*/false, b,
      /*c_mod=*/(short)0, c, /*reuse_a=*/false, /*reuse_b=*/false);
}

static __device__ __forceinline__ float fast_sigmoid(float a) {
  // v_rcp_f32 instead of the v_div_scale/fmas/fixup chain.
  return __builtin_amdgcn_rcpf(1.f + __expf(-a));
}

// ---------------------------------------------------------------------------
// K0: f32 -> f16 weight staging into workspace (L2-resident afterwards).
//   WLH[r*512 + c]              = l_reduce_w[r][c]
//   W3H[(tap*64 + o)*64 + i]    = conv3_w[o][i][tap]   (tap-major for B-frags)
//   W5H[(tap*64 + o)*64 + i]    = conv5_w[o][i][tap]
// ---------------------------------------------------------------------------
__global__ void prep_weights_kernel(const float* __restrict__ wl,
                                    const float* __restrict__ w3,
                                    const float* __restrict__ w5,
                                    _Float16* __restrict__ wlh,
                                    _Float16* __restrict__ w3h,
                                    _Float16* __restrict__ w5h) {
  int i = blockIdx.x * blockDim.x + threadIdx.x;
  if (i < R_N * C_N) wlh[i] = (_Float16)wl[i];
  if (i < 9 * 64 * 64) {
    int tap = i >> 12, o = (i >> 6) & 63, ic = i & 63;
    w3h[i] = (_Float16)w3[(o * 64 + ic) * 9 + tap];
  }
  if (i < 25 * 64 * 64) {
    int tap = i >> 12, o = (i >> 6) & 63, ic = i & 63;
    w5h[i] = (_Float16)w5[(o * 64 + ic) * 25 + tap];
  }
}

// ---------------------------------------------------------------------------
// LDS layout (bytes):
//   Xh    [64][520] f16   @ 0        (66560)  x[b] in f16, rows 49..63 zero
//   XLh   [64][136] f16   @ 66560    (17408)  l-reduce output (pos, chan)
//   XLs   [64][132] f32   @ 83968    (33792)  conv out after bias+star
//   xmaxs [512]     f32   @ 117760
//   xavgs [512]     f32   @ 119808
//   g1s   [128]     f32   @ 121856
//   g2s   [128]     f32   @ 122368
//   cagg  [512]     f32   @ 122880
//   sval  [64]      f32   @ 124928
//   sgp   [1]       f32   @ 125184
// 125440 B total -> 2 workgroups per 320 KB WGP.
// ---------------------------------------------------------------------------
#define SMEM_BYTES 125440

__global__ __launch_bounds__(256) void mafn_main_kernel(
    const float* __restrict__ x,
    const float* __restrict__ g_reduce_w,
    const float* __restrict__ conv3_b,
    const float* __restrict__ conv5_b,
    const float* __restrict__ channel_w,
    const float* __restrict__ channel_b,
    const float* __restrict__ spatial_w,
    const float* __restrict__ spatial_b,
    const float* __restrict__ star_scale,
    const float* __restrict__ star_bias,
    const _Float16* __restrict__ WLH,
    const _Float16* __restrict__ W3H,
    const _Float16* __restrict__ W5H,
    float* __restrict__ out) {
  extern __shared__ char smem[];
  _Float16* Xh   = (_Float16*)(smem);
  _Float16* XLh  = (_Float16*)(smem + 66560);
  float* XLs     = (float*)(smem + 83968);
  float* xmaxs   = (float*)(smem + 117760);
  float* xavgs   = (float*)(smem + 119808);
  float* g1s     = (float*)(smem + 121856);
  float* g2s     = (float*)(smem + 122368);
  float* cagg    = (float*)(smem + 122880);
  float* sval    = (float*)(smem + 124928);
  float* sgp     = (float*)(smem + 125184);

  const int tid = threadIdx.x;
  const int b = blockIdx.x;
  const float sscale = star_scale[0];
  const float sbias = star_bias[0];
  const float* xb = x + (size_t)b * (L_N * C_N);

  // ---- Phase A: stream x[b] with b128 loads -> stats + f16 LDS copy ----
  // 128 column-groups of 4; even/odd rows split across the two thread halves,
  // partials combined through LDS.
  {
    const int cg = (tid & 127) * 4;  // column group base
    const int rp = tid >> 7;         // 0: even rows, 1: odd rows
    floatv4 vmax, vsum;
#pragma unroll
    for (int e = 0; e < 4; ++e) { vmax[e] = -3.4e38f; vsum[e] = 0.f; }

    for (int l = rp; l < L_N; l += 2) {
      floatv4 v = *(const floatv4*)(xb + l * C_N + cg);  // global_load_b128
#pragma unroll
      for (int e = 0; e < 4; ++e) {
        vmax[e] = fmaxf(vmax[e], v[e]);
        vsum[e] += v[e];
      }
      *(half4*)(Xh + l * 520 + cg) = __builtin_convertvector(v, half4);
    }
    if (rp == 1) {  // publish odd-row partials
      *(floatv4*)(xmaxs + cg) = vmax;
      *(floatv4*)(xavgs + cg) = vsum;
    }
    __syncthreads();
    if (rp == 0) {  // combine + finalize
      floatv4 om = *(floatv4*)(xmaxs + cg);
      floatv4 os = *(floatv4*)(xavgs + cg);
#pragma unroll
      for (int e = 0; e < 4; ++e) {
        om[e] = fmaxf(om[e], vmax[e]);
        os[e] = (os[e] + vsum[e]) * (1.f / 49.f);
      }
      *(floatv4*)(xmaxs + cg) = om;
      *(floatv4*)(xavgs + cg) = os;
    }
    // zero pad rows 49..63 of Xh (ds_store_b64)
    half4 z;
#pragma unroll
    for (int e = 0; e < 4; ++e) z[e] = (_Float16)0.f;
    for (int i = tid; i < 15 * 130; i += 256)
      *(half4*)(Xh + 49 * 520 + i * 4) = z;
  }
  __syncthreads();

  // ---- Phase B: g1/g2 = star_relu(stats @ g_reduce_w^T) ----------------
  if (tid < R_N) {
    const float* gw = g_reduce_w + tid * C_N;
    float a1 = 0.f, a2 = 0.f;
    for (int c = 0; c < C_N; ++c) {
      float w = gw[c];
      a1 += xmaxs[c] * w;
      a2 += xavgs[c] * w;
    }
    float r1 = fmaxf(a1, 0.f), r2 = fmaxf(a2, 0.f);
    g1s[tid] = sscale * r1 * r1 + sbias;
    g2s[tid] = sscale * r2 * r2 + sbias;
  }
  __syncthreads();

  // ---- Phase C: spatial-gate scalar + channel aggregation --------------
  if (tid == 0) {
    float s = 0.f;
    for (int r = 0; r < R_N; ++r)
      s += g1s[r] * spatial_w[128 + r] + g2s[r] * spatial_w[256 + r];
    sgp[0] = s + spatial_b[0];
  }
  for (int j = 0; j < 2; ++j) {
    int c = tid + 256 * j;
    const float* cw = channel_w + c * R_N;
    float a = 0.f;
    for (int r = 0; r < R_N; ++r) a += (g1s[r] + g2s[r]) * cw[r];
    a += 2.f * channel_b[c];
    cagg[c] = fast_sigmoid(a);
  }
  // cagg/sgp consumed only after later barriers; GEMM below needs only Xh.

  // ---- WMMA lane decomposition (wave32) --------------------------------
  const int wv = tid >> 5;          // wave id 0..7
  const int lane = tid & 31;
  const int nrow = lane & 15;       // A-row (M) / B-col (N) within tile
  const int hihalf = lane >> 4;
  const int kbA = hihalf ? 8 : 0;   // A frag: hi lanes take k+8 / k+24 chunks
  const int kbB = hihalf ? 8 : 0;   // B frag: hi lanes take k+8 / k+24 chunks

  float8 zero8;
#pragma unroll
  for (int e = 0; e < 8; ++e) zero8[e] = 0.f;

  // ---- Phase D: XLpre(64x128) = Xh(64x512) @ l_reduce_w^T(512x128) -----
  {
    float8 acc[4];
#pragma unroll
    for (int m = 0; m < 4; ++m) acc[m] = zero8;

    const _Float16* wrow = WLH + (wv * 16 + nrow) * C_N;  // B col = channel r
    for (int ks = 0; ks < 16; ++ks) {  // K = 512 in steps of 32
      const _Float16* bp = wrow + ks * 32 + kbB;
      half16 bf = cat8(*(const half8*)bp, *(const half8*)(bp + 16));
#pragma unroll
      for (int m = 0; m < 4; ++m) {
        const _Float16* ap = Xh + (m * 16 + nrow) * 520 + ks * 32 + kbA;
        half16 af = cat8(*(const half8*)ap, *(const half8*)(ap + 16));
        acc[m] = wmma_f16(af, bf, acc[m]);
      }
    }
    // Scatter D tile to XLh[pos][chan]; rows l>=49 are exact zeros.
#pragma unroll
    for (int m = 0; m < 4; ++m)
#pragma unroll
      for (int i = 0; i < 8; ++i) {
        int l = m * 16 + i + (hihalf ? 8 : 0);
        XLh[l * 136 + wv * 16 + nrow] = (_Float16)acc[m][i];
      }
  }
  __syncthreads();

  // ---- Phase E: convs as tap-decomposed WMMA GEMMs ---------------------
  {
    const bool is5 = wv >= 4;            // waves 0-3: conv3, 4-7: conv5
    const int ntile = wv & 3;
    const int rad = is5 ? 2 : 1;
    const int ntaps = is5 ? 25 : 9;
    const int ksz = 2 * rad + 1;
    const int cbase = is5 ? 64 : 0;      // input-channel base (and out offset)
    const _Float16* WB = is5 ? W5H : W3H;
    const int ocol = ntile * 16 + nrow;  // output channel within this conv
    const float bias = (is5 ? conv5_b : conv3_b)[ocol];

    int hp[4], wp[4];
    bool pv[4];
#pragma unroll
    for (int m = 0; m < 4; ++m) {
      int p = m * 16 + nrow;
      pv[m] = p < L_N;
      hp[m] = p / 7;
      wp[m] = p % 7;
    }

    float8 acc[4];
#pragma unroll
    for (int m = 0; m < 4; ++m) acc[m] = zero8;

    for (int t = 0; t < ntaps; ++t) {
      int kh = t / ksz - rad, kw = t % ksz - rad;
      int q[4];
#pragma unroll
      for (int m = 0; m < 4; ++m) {
        int hi2 = hp[m] + kh, wi2 = wp[m] + kw;
        bool ok = pv[m] && hi2 >= 0 && hi2 < 7 && wi2 >= 0 && wi2 < 7;
        q[m] = ok ? (hi2 * 7 + wi2) : 56;  // row 56 is guaranteed zero
      }
#pragma unroll
      for (int ks = 0; ks < 2; ++ks) {  // K = 64 in steps of 32
        const _Float16* bp = WB + (t * 64 + ocol) * 64 + ks * 32 + kbB;
        half16 bf = cat8(*(const half8*)bp, *(const half8*)(bp + 16));
#pragma unroll
        for (int m = 0; m < 4; ++m) {
          const _Float16* ap = XLh + q[m] * 136 + cbase + ks * 32 + kbA;
          half16 af = cat8(*(const half8*)ap, *(const half8*)(ap + 16));
          acc[m] = wmma_f16(af, bf, acc[m]);
        }
      }
    }
    const int rfull = cbase + ocol;  // channel index in concatenated output
#pragma unroll
    for (int m = 0; m < 4; ++m)
#pragma unroll
      for (int i = 0; i < 8; ++i) {
        float v = acc[m][i] + bias;
        v = fmaxf(v, 0.f);
        v = sscale * v * v + sbias;
        int l = m * 16 + i + (hihalf ? 8 : 0);
        XLs[l * 132 + rfull] = v;
      }
  }
  __syncthreads();

  // ---- Phase F: spatial gate per position ------------------------------
  if (tid < L_N) {
    float s = 0.f;
    for (int r = 0; r < R_N; ++r) s += XLs[tid * 132 + r] * spatial_w[r];
    sval[tid] = fast_sigmoid(s + sgp[0]);
  }
  __syncthreads();

  // ---- Phase G: out = x * c_aggr * s_aggr (b128 in / b128 out) ---------
  float* ob = out + (size_t)b * (L_N * C_N);
  for (int i = tid; i < (L_N * C_N) / 4; i += 256) {
    int base = i * 4;
    int l = base >> 9;             // 4 consecutive c share one row l
    int c = base & (C_N - 1);
    floatv4 xv = *(const floatv4*)(xb + base);
    floatv4 cg4 = *(const floatv4*)(cagg + c);
    float sv = sval[l];
    floatv4 o;
#pragma unroll
    for (int e = 0; e < 4; ++e) o[e] = xv[e] * cg4[e] * sv;
    *(floatv4*)(ob + base) = o;
  }
}

// ---------------------------------------------------------------------------
extern "C" void kernel_launch(void* const* d_in, const int* in_sizes, int n_in,
                              void* d_out, int out_size, void* d_ws,
                              size_t ws_size, hipStream_t stream) {
  (void)in_sizes; (void)n_in; (void)out_size; (void)ws_size;

  const float* x          = (const float*)d_in[0];
  const float* g_reduce_w = (const float*)d_in[1];
  const float* l_reduce_w = (const float*)d_in[2];
  const float* conv3_w    = (const float*)d_in[3];
  const float* conv3_b    = (const float*)d_in[4];
  const float* conv5_w    = (const float*)d_in[5];
  const float* conv5_b    = (const float*)d_in[6];
  const float* channel_w  = (const float*)d_in[7];
  const float* channel_b  = (const float*)d_in[8];
  const float* spatial_w  = (const float*)d_in[9];
  const float* spatial_b  = (const float*)d_in[10];
  const float* star_scale = (const float*)d_in[11];
  const float* star_bias  = (const float*)d_in[12];

  char* wsb = (char*)d_ws;
  _Float16* WLH = (_Float16*)(wsb);            // 128*512 f16 = 131072 B
  _Float16* W3H = (_Float16*)(wsb + 131072);   //  9*64*64 f16 =  73728 B
  _Float16* W5H = (_Float16*)(wsb + 204800);   // 25*64*64 f16 = 204800 B

  prep_weights_kernel<<<(25 * 64 * 64 + 255) / 256, 256, 0, stream>>>(
      l_reduce_w, conv3_w, conv5_w, WLH, W3H, W5H);

  static bool attr_set = false;  // host-side, deterministic, capture-safe
  if (!attr_set) {
    hipFuncSetAttribute((const void*)mafn_main_kernel,
                        hipFuncAttributeMaxDynamicSharedMemorySize, SMEM_BYTES);
    attr_set = true;
  }

  mafn_main_kernel<<<B_N, 256, SMEM_BYTES, stream>>>(
      x, g_reduce_w, conv3_b, conv5_b, channel_w, channel_b, spatial_w,
      spatial_b, star_scale, star_bias, WLH, W3H, W5H, (float*)d_out);
}
